// SpikeModulatedAttentionBlock_21629455303096
// MI455X (gfx1250) — matrix operand order repaired and checked by reading, not compiled, not run on hardware
//
#include <hip/hip_runtime.h>
#include <hip/hip_bf16.h>
#include <math.h>

typedef __attribute__((ext_vector_type(16))) _Float16 v16h;
typedef __attribute__((ext_vector_type(8)))  float    v8f;
typedef __attribute__((ext_vector_type(4)))  unsigned int u32x4;
typedef __attribute__((ext_vector_type(8)))  int      i32x8;
typedef __attribute__((ext_vector_type(4)))  int      i32x4;

#define Bsz 4
#define LQ  2048
#define LK  2048
#define ND  512
#define NH  8
#define DS  64
#define GK  512   // inner dim of all projection GEMMs

// ---------------------------------------------------------------------------
// TDM: issue a 2D tensor_load_to_lds. D# layout per CDNA5 ISA ch.8:
//  group0: [1:0]=count=1, [63:32]=lds_addr, [120:64]=global_addr, [127:126]=2
//  group1: [17:16]=data_size(1->2B), [79:48]=tensor_dim0, [111:80]=tensor_dim1,
//          [127:112]=tile_dim0, [143:128]=tile_dim1, [207:160]=dim0_stride
// ---------------------------------------------------------------------------
__device__ __forceinline__ void tdm_load_2d_f16(
    unsigned int lds_addr, const _Float16* gptr,
    unsigned int tensor_d0, unsigned int tensor_d1,
    unsigned int tile_d0, unsigned int tile_d1, unsigned int stride0) {
  unsigned long long ga = (unsigned long long)(uintptr_t)gptr;
  u32x4 g0;
  g0[0] = 1u;                                               // count = 1
  g0[1] = lds_addr;                                         // LDS byte address
  g0[2] = (unsigned int)ga;                                 // global_addr[31:0]
  g0[3] = (unsigned int)((ga >> 32) & 0x1FFFFFFu) | (2u << 30);  // [56:32]|type=2
  i32x8 g1;
  g1[0] = (int)(1u << 16);                                  // data_size = 2 bytes
  g1[1] = (int)((tensor_d0 & 0xFFFFu) << 16);               // tensor_dim0[15:0]
  g1[2] = (int)((tensor_d0 >> 16) | ((tensor_d1 & 0xFFFFu) << 16));
  g1[3] = (int)((tensor_d1 >> 16) | (tile_d0 << 16));       // tile_dim0
  g1[4] = (int)(tile_d1 & 0xFFFFu);                         // tile_dim1 (tile_dim2=0)
  g1[5] = (int)stride0;                                     // dim0_stride[31:0]
  g1[6] = 0;
  g1[7] = 0;
  i32x4 z4 = {0, 0, 0, 0};
#if __clang_major__ >= 23
  i32x8 z8 = {0, 0, 0, 0, 0, 0, 0, 0};
  __builtin_amdgcn_tensor_load_to_lds(g0, g1, z4, z4, z8, 0);
#else
  __builtin_amdgcn_tensor_load_to_lds(g0, g1, z4, z4, 0);
#endif
}

// ---------------------------------------------------------------------------
// f16 GEMM: C[M,512] = A16[M,512] * Wt16[512n,512k]^T + bias.
// 256 threads = 8 waves (4 in M x 2 in N); WG tile 128x64; each wave owns a
// 32x32 output (2x2 WMMA tiles, fragments reused twice). Tiles staged into
// LDS by the Tensor Data Mover; wave0 issues TDM + s_wait_tensorcnt.
// ---------------------------------------------------------------------------
__global__ __launch_bounds__(256) void smab_gemm_kernel(
    const _Float16* __restrict__ A16, const _Float16* __restrict__ Wt16,
    const float* __restrict__ bias, float* __restrict__ C, int Mrows) {
  __shared__ _Float16 sA[128 * 32];   // [row][k]
  __shared__ _Float16 sBt[64 * 32];   // [n][k]

  const int m0   = blockIdx.x * 128;
  const int n0   = blockIdx.y * 64;
  const int tid  = threadIdx.x;
  const int wave = tid >> 5;
  const int lane = tid & 31;
  const int wm   = wave >> 1;   // 0..3 -> 32-row band
  const int wn   = wave & 1;    // 0..1 -> 32-col band
  const int half = lane >> 4;
  const int l16  = lane & 15;

  const unsigned int ldsA = (unsigned int)(uintptr_t)&sA[0];
  const unsigned int ldsB = (unsigned int)(uintptr_t)&sBt[0];

  v8f acc[2][2] = {{{}, {}}, {{}, {}}};

  for (int kb = 0; kb < GK; kb += 32) {
    if (wave == 0) {
      // A tile: 128 rows x 32 k, row-major; B tile: 64 n x 32 k, row-major
      tdm_load_2d_f16(ldsA, A16 + (size_t)m0 * GK + kb, GK, (unsigned)Mrows,
                      32u, 128u, GK);
      tdm_load_2d_f16(ldsB, Wt16 + (size_t)n0 * GK + kb, GK, ND,
                      32u, 64u, GK);
      __builtin_amdgcn_s_wait_tensorcnt(0);
    }
    __syncthreads();

    v16h a[2], b[2];
    #pragma unroll
    for (int i = 0; i < 2; ++i) {
      const int arow = wm * 32 + i * 16 + l16;
      #pragma unroll
      for (int e = 0; e < 8; ++e) {
        a[i][e]     = sA[arow * 32 + half * 16 + e];
        a[i][e + 8] = sA[arow * 32 + half * 16 + 8 + e];
      }
      const int bcol = wn * 32 + i * 16 + l16;
      #pragma unroll
      for (int e = 0; e < 16; ++e) b[i][e] = sBt[bcol * 32 + half * 16 + e];
    }
    #pragma unroll
    for (int i = 0; i < 2; ++i)
      #pragma unroll
      for (int j = 0; j < 2; ++j)
        acc[i][j] = __builtin_amdgcn_wmma_f32_16x16x32_f16(
            false, a[i], false, b[j], (short)0, acc[i][j], false, false);
    __syncthreads();
  }

  // C layout per tile: lane -> col, VGPR r -> row r + 8*half
  #pragma unroll
  for (int j = 0; j < 2; ++j) {
    const int ccol = n0 + wn * 32 + j * 16 + l16;
    const float bval = bias[ccol];
    #pragma unroll
    for (int i = 0; i < 2; ++i) {
      #pragma unroll
      for (int r = 0; r < 8; ++r) {
        const int crow = m0 + wm * 32 + i * 16 + r + 8 * half;
        C[(size_t)crow * ND + ccol] = acc[i][j][r] + bval;
      }
    }
  }
}

// ---------------------------------------------------------------------------
// Packs: f32 -> f16 elementwise; f32 W[k][n] -> f16 Wt[n][k] (transposed).
// ---------------------------------------------------------------------------
__global__ __launch_bounds__(256) void smab_packa_kernel(
    const float* __restrict__ A, _Float16* __restrict__ A16) {
  size_t i = (size_t)blockIdx.x * 256 + threadIdx.x;
  A16[i] = (_Float16)A[i];
}

__global__ __launch_bounds__(256) void smab_packwt_kernel(
    const float* __restrict__ W, _Float16* __restrict__ Wt) {
  int i = blockIdx.x * 256 + threadIdx.x;   // over 512*512, coalesced in n
  int k = i >> 9, n = i & 511;
  Wt[(size_t)n * GK + k] = (_Float16)W[i];
}

// kf16 = (f16)k (same layout); vt16[b][h][d][key] = (f16)v[b][key][h*64+d]
__global__ __launch_bounds__(256) void smab_packkv_kernel(
    const float* __restrict__ k, const float* __restrict__ v,
    _Float16* __restrict__ kf16, _Float16* __restrict__ vt16) {
  size_t i = (size_t)blockIdx.x * 256 + threadIdx.x;  // over B*LK*ND
  kf16[i] = (_Float16)k[i];
  size_t b   = i / ((size_t)LK * ND);
  size_t rem = i % ((size_t)LK * ND);
  size_t key = rem / ND;
  size_t c   = rem % ND;
  size_t h = c / DS, d = c % DS;
  vt16[(((b * NH + h) * DS) + d) * LK + key] = (_Float16)v[i];
}

// ---------------------------------------------------------------------------
// Flash attention + residual: O = qh + softmax(mask(qk^T/sqrt(ND)/temp)) @ vh
// 1 wave per (b, h, 16-query tile). 128 threads = 4 waves per block.
// ---------------------------------------------------------------------------
__global__ __launch_bounds__(128) void smab_attn_kernel(
    const float* __restrict__ q, const _Float16* __restrict__ kf16,
    const _Float16* __restrict__ vt16, const int* __restrict__ mask,
    const float* __restrict__ temp, float* __restrict__ O) {
  __shared__ _Float16 sP[4][16 * 32];  // per-wave P-tile transpose scratch

  const int tid  = threadIdx.x;
  const int wave = tid >> 5;
  const int lane = tid & 31;
  const int half = lane >> 4;
  const int l16  = lane & 15;

  const int gw = blockIdx.x * 4 + wave;       // 0 .. B*NH*(LQ/16)-1
  const int qt = gw & 127;
  const int h  = (gw >> 7) & 7;
  const int b  = gw >> 10;
  const int q0 = qt * 16;

  const float*    qb  = q    + (size_t)b * LQ * ND;
  const _Float16* kfb = kf16 + (size_t)b * LK * ND;
  const _Float16* vtb = vt16 + ((size_t)b * NH + h) * DS * LK;
  const int*      mb  = mask + (size_t)b * LQ * LK;
  const float*    tb  = temp + (size_t)b * LK;

  v16h fa[2];
  #pragma unroll
  for (int kb = 0; kb < 2; ++kb) {
    const float* p0 = qb + (size_t)(q0 + l16) * ND + h * DS + kb * 32 + half * 16;
    #pragma unroll
    for (int i = 0; i < 8; ++i) {
      fa[kb][i]     = (_Float16)p0[i];
      fa[kb][i + 8] = (_Float16)p0[8 + i];
    }
  }

  float Mx[8], L[8];
  v8f oa[4] = {{}, {}, {}, {}};
  #pragma unroll
  for (int r = 0; r < 8; ++r) { Mx[r] = -1e30f; L[r] = 0.0f; }

  const float inv_sqrt_nd = 0.044194173824159216f;  // 1/sqrt(512)
  _Float16* sp = sP[wave];

  for (int kc = 0; kc < LK; kc += 32) {
    float s[2][8];
    #pragma unroll
    for (int t = 0; t < 2; ++t) {
      const int key = kc + t * 16 + l16;
      v16h fb0, fb1;
      const _Float16* kp = kfb + (size_t)key * ND + h * DS + half * 16;
      #pragma unroll
      for (int i = 0; i < 16; ++i) { fb0[i] = kp[i]; fb1[i] = kp[32 + i]; }
      v8f sc = {};
      sc = __builtin_amdgcn_wmma_f32_16x16x32_f16(false, fa[0], false, fb0,
                                                  (short)0, sc, false, false);
      sc = __builtin_amdgcn_wmma_f32_16x16x32_f16(false, fa[1], false, fb1,
                                                  (short)0, sc, false, false);
      const float scale = inv_sqrt_nd / tb[key];
      #pragma unroll
      for (int r = 0; r < 8; ++r) {
        const int row = q0 + r + 8 * half;
        const int mk = mb[(size_t)row * LK + key];
        s[t][r] = mk ? sc[r] * scale : -1e9f;
      }
    }

    float mnew[8];
    #pragma unroll
    for (int r = 0; r < 8; ++r) {
      float m = fmaxf(s[0][r], s[1][r]);
      #pragma unroll
      for (int x = 8; x > 0; x >>= 1) m = fmaxf(m, __shfl_xor(m, x, 32));
      mnew[r] = fmaxf(Mx[r], m);
    }
    #pragma unroll
    for (int r = 0; r < 8; ++r) {
      const float f = __expf(Mx[r] - mnew[r]);
      L[r] *= f;
      #pragma unroll
      for (int j = 0; j < 4; ++j) oa[j][r] *= f;
      Mx[r] = mnew[r];
    }
    #pragma unroll
    for (int t = 0; t < 2; ++t) {
      #pragma unroll
      for (int r = 0; r < 8; ++r) {
        const float p = __expf(s[t][r] - Mx[r]);
        float ps = p;
        #pragma unroll
        for (int x = 8; x > 0; x >>= 1) ps += __shfl_xor(ps, x, 32);
        L[r] += ps;
        sp[(r + 8 * half) * 32 + t * 16 + l16] = (_Float16)p;
      }
    }
    __syncthreads();

    v16h pf;
    #pragma unroll
    for (int i = 0; i < 8; ++i) {
      pf[i]     = sp[l16 * 32 + half * 16 + i];
      pf[i + 8] = sp[l16 * 32 + half * 16 + 8 + i];
    }
    __syncthreads();

    if (kc + 32 < LK)
      __builtin_prefetch(kfb + (size_t)(kc + 32 + l16) * ND + h * DS, 0, 0);

    #pragma unroll
    for (int j = 0; j < 4; ++j) {
      v16h fv;
      const _Float16* vp = vtb + (size_t)(j * 16 + l16) * LK + kc + half * 16;
      #pragma unroll
      for (int i = 0; i < 16; ++i) fv[i] = vp[i];
      oa[j] = __builtin_amdgcn_wmma_f32_16x16x32_f16(false, pf, false, fv,
                                                     (short)0, oa[j], false, false);
    }
  }

  float* Ob = O + (size_t)b * LQ * ND;
  #pragma unroll
  for (int j = 0; j < 4; ++j) {
    const int col = h * DS + j * 16 + l16;
    #pragma unroll
    for (int r = 0; r < 8; ++r) {
      const int row = q0 + r + 8 * half;
      const float qres = qb[(size_t)row * ND + col];
      Ob[(size_t)row * ND + col] = qres + oa[j][r] / L[r];
    }
  }
}

// ---------------------------------------------------------------------------
// LayerNorm over 512 channels; if Y != nullptr computes LN(X + relu(Y)).
// ---------------------------------------------------------------------------
__global__ __launch_bounds__(256) void smab_ln_kernel(
    const float* __restrict__ X, const float* __restrict__ Y,
    const float* __restrict__ g, const float* __restrict__ bt,
    float* __restrict__ out) {
  __shared__ float ss[8], sq[8];
  const int tid = threadIdx.x;
  const size_t row = blockIdx.x;
  const float* x = X + row * ND;
  float a0 = x[tid], a1 = x[tid + 256];
  if (Y) {
    const float* y = Y + row * ND;
    a0 += fmaxf(y[tid], 0.0f);
    a1 += fmaxf(y[tid + 256], 0.0f);
  }
  float s = a0 + a1, q2 = a0 * a0 + a1 * a1;
  #pragma unroll
  for (int m = 16; m > 0; m >>= 1) {
    s  += __shfl_xor(s,  m, 32);
    q2 += __shfl_xor(q2, m, 32);
  }
  const int wave = tid >> 5, lane = tid & 31;
  if (lane == 0) { ss[wave] = s; sq[wave] = q2; }
  __syncthreads();
  float S = 0.0f, Q2 = 0.0f;
  #pragma unroll
  for (int w = 0; w < 8; ++w) { S += ss[w]; Q2 += sq[w]; }
  const float mean = S * (1.0f / ND);
  const float var  = Q2 * (1.0f / ND) - mean * mean;
  const float inv  = rsqrtf(var + 1e-5f);
  out[row * ND + tid]       = (a0 - mean) * inv * g[tid]       + bt[tid];
  out[row * ND + tid + 256] = (a1 - mean) * inv * g[tid + 256] + bt[tid + 256];
}

// ---------------------------------------------------------------------------
extern "C" void kernel_launch(void* const* d_in, const int* in_sizes, int n_in,
                              void* d_out, int out_size, void* d_ws, size_t ws_size,
                              hipStream_t stream) {
  const float* Q    = (const float*)d_in[0];
  const float* K    = (const float*)d_in[1];
  const int*   mask = (const int*)d_in[2];
  const float* temp = (const float*)d_in[3];
  const float* Wq   = (const float*)d_in[4];
  const float* bq   = (const float*)d_in[5];
  const float* Wk   = (const float*)d_in[6];
  const float* bk   = (const float*)d_in[7];
  const float* Wv   = (const float*)d_in[8];
  const float* bv   = (const float*)d_in[9];
  const float* Wo   = (const float*)d_in[10];
  const float* bo   = (const float*)d_in[11];
  const float* g0   = (const float*)d_in[12];
  const float* b0   = (const float*)d_in[13];
  const float* g1   = (const float*)d_in[14];
  const float* b1   = (const float*)d_in[15];
  float* out = (float*)d_out;

  const size_t SZ = (size_t)Bsz * LQ * ND;   // 4,194,304 elements
  const size_t WSZ = (size_t)GK * ND;        // 262,144 elements per weight
  char* ws = (char*)d_ws;
  float*    qbuf = (float*)(ws + 0 * SZ * 4);
  float*    kbuf = (float*)(ws + 1 * SZ * 4);
  float*    vbuf = (float*)(ws + 2 * SZ * 4);
  float*    Oatt = (float*)(ws + 3 * SZ * 4);
  float*    Xbuf = (float*)(ws + 4 * SZ * 4);
  float*    Ybuf = (float*)(ws + 5 * SZ * 4);
  char* h16 = ws + 6 * SZ * 4;
  _Float16* kf16  = (_Float16*)(h16 + 0 * SZ * 2);
  _Float16* vt16  = (_Float16*)(h16 + 1 * SZ * 2);
  _Float16* Qf16  = (_Float16*)(h16 + 2 * SZ * 2);
  _Float16* Kf16  = (_Float16*)(h16 + 3 * SZ * 2);
  _Float16* Xf16  = (_Float16*)(h16 + 4 * SZ * 2);
  _Float16* WtQ   = (_Float16*)(h16 + 5 * SZ * 2);
  _Float16* WtK   = WtQ + WSZ;
  _Float16* WtV   = WtK + WSZ;
  _Float16* WtO   = WtV + WSZ;

  const int M = Bsz * LQ;                       // 8192
  const dim3 gGemm(M / 128, ND / 64);           // 64 x 8
  const unsigned gPackA = (unsigned)(SZ / 256);
  const unsigned gPackW = (unsigned)(WSZ / 256);

  // 0: pack operands to f16 (A row-major, W transposed [n][k])
  smab_packwt_kernel<<<gPackW, 256, 0, stream>>>(Wq, WtQ);
  smab_packwt_kernel<<<gPackW, 256, 0, stream>>>(Wk, WtK);
  smab_packwt_kernel<<<gPackW, 256, 0, stream>>>(Wv, WtV);
  smab_packwt_kernel<<<gPackW, 256, 0, stream>>>(Wo, WtO);
  smab_packa_kernel<<<gPackA, 256, 0, stream>>>(Q, Qf16);
  smab_packa_kernel<<<gPackA, 256, 0, stream>>>(K, Kf16);
  // 1-3: projections (TDM-staged WMMA GEMM)
  smab_gemm_kernel<<<gGemm, 256, 0, stream>>>(Qf16, WtQ, bq, qbuf, M);
  smab_gemm_kernel<<<gGemm, 256, 0, stream>>>(Kf16, WtK, bk, kbuf, M);
  smab_gemm_kernel<<<gGemm, 256, 0, stream>>>(Kf16, WtV, bv, vbuf, M);
  // 4: pack projected K/V for attention (V transposed per head)
  smab_packkv_kernel<<<gPackA, 256, 0, stream>>>(kbuf, vbuf, kf16, vt16);
  // 5: flash attention + residual
  smab_attn_kernel<<<(Bsz * NH * (LQ / 16)) / 4, 128, 0, stream>>>(
      qbuf, kf16, vt16, mask, temp, Oatt);
  // 6: LN0
  smab_ln_kernel<<<M, 256, 0, stream>>>(Oatt, nullptr, g0, b0, Xbuf);
  // 7: output projection
  smab_packa_kernel<<<gPackA, 256, 0, stream>>>(Xbuf, Xf16);
  smab_gemm_kernel<<<gGemm, 256, 0, stream>>>(Xf16, WtO, bo, Ybuf, M);
  // 8: X + relu(Y), LN1 -> out
  smab_ln_kernel<<<M, 256, 0, stream>>>(Xbuf, Ybuf, g1, b1, out);
}
